// RNNModel_49048526520845
// MI455X (gfx1250) — matrix operand order repaired and checked by reading, not compiled, not run on hardware
//
#include <hip/hip_runtime.h>

// ---------------------------------------------------------------------------
// Types for CDNA5 WMMA bf16 path: D(f32 16x16) = A(bf16 16x32) x B(bf16 32x16) + C
// ---------------------------------------------------------------------------
typedef __attribute__((ext_vector_type(16))) __bf16 v16bf;
typedef __attribute__((ext_vector_type(8)))  __bf16 v8bf;
typedef __attribute__((ext_vector_type(8)))  float  v8f;

__device__ __forceinline__ unsigned short f2bf_bits(float f) {
  union { float f; unsigned u; } v; v.f = f;
  unsigned r = v.u + 0x7FFFu + ((v.u >> 16) & 1u);   // round-to-nearest-even
  return (unsigned short)(r >> 16);
}
__device__ __forceinline__ float bf2f(unsigned short h) {
  union { unsigned u; float f; } v; v.u = ((unsigned)h) << 16;
  return v.f;
}

// A-fragment (16x32 bf16, row-major source, lda elements) from LDS.
// This code's contiguous k order (same as B): lane<16 -> M=lane, K={0..15}; lane>=16 -> K={16..31}.
__device__ __forceinline__ v16bf load_a_lds(const unsigned short* base, int lda,
                                            int k0, int lane) {
  const int m  = lane & 15;
  const int kh = lane >> 4;
  const v8bf lo = *reinterpret_cast<const v8bf*>(base + m * lda + k0 + kh * 16);
  const v8bf hi = *reinterpret_cast<const v8bf*>(base + m * lda + k0 + 8 + kh * 16);
  union { v16bf v; v8bf h[2]; } u;
  u.h[0] = lo; u.h[1] = hi;
  return u.v;
}

// B-fragment (32x16 bf16) from an N-major (transposed) weight matrix bt[N][K].
// This code's contiguous k order (same as A): lane = N + 16*(K-half); each lane owns 16 contiguous K values.
__device__ __forceinline__ v16bf load_b_global(const unsigned short* bt, int K,
                                               int col0, int k0, int lane) {
  const int n  = lane & 15;
  const int kg = lane >> 4;
  return *reinterpret_cast<const v16bf*>(bt + (size_t)(col0 + n) * K + k0 + kg * 16);
}

__device__ __forceinline__ v8f wmma_bf16(v16bf a, v16bf b, v8f c) {
  return __builtin_amdgcn_wmma_f32_16x16x32_bf16(false, a, false, b,
                                                 (short)0, c, false, false);
}

// ---------------------------------------------------------------------------
// Problem constants
// ---------------------------------------------------------------------------
#define BB   256
#define TT   512
#define FF   256      // W*C_IN
#define HH   512
#define GG   1536     // 3*H
#define BT   (BB * TT)

// ---------------------------------------------------------------------------
// Kernel 0: transpose f32 [K][N] -> bf16 [N][K] (weights -> N-major for B-frag)
// ---------------------------------------------------------------------------
__global__ void k_transpose_bf16(const float* __restrict__ in,
                                 unsigned short* __restrict__ out,
                                 int K, int N) {
  int idx = blockIdx.x * 256 + threadIdx.x;
  if (idx >= K * N) return;
  int n = idx / K, k = idx - n * K;
  out[idx] = f2bf_bits(in[(size_t)k * N + n]);
}

// ---------------------------------------------------------------------------
// Kernel 1: fused  mx = relu(X @ W1 + b1) @ Wx + bx   (X = [BT][256] f32)
// 32 rows per workgroup; activations stay in LDS (never touch HBM).
// 8 waves: wm = wave&1 selects M-tile, wn = wave>>1 selects N-group of 4.
// ---------------------------------------------------------------------------
__global__ __launch_bounds__(256) void k_fused_proj(
    const float* __restrict__ x_in,          // [BT][256]
    const unsigned short* __restrict__ w1t,  // [512][256] bf16
    const float* __restrict__ b1,            // [512]
    const unsigned short* __restrict__ wxt,  // [1536][512] bf16
    const float* __restrict__ bx,            // [1536]
    unsigned short* __restrict__ mx)         // [BT][1536] bf16
{
  __shared__ __align__(32) unsigned short a_lds[32 * FF];   // 16 KB
  __shared__ __align__(32) unsigned short xs_lds[32 * HH];  // 32 KB

  const int tid  = threadIdx.x;
  const int lane = tid & 31;
  const int wave = tid >> 5;
  const int wm   = wave & 1;
  const int wn   = wave >> 1;
  const size_t row0 = (size_t)blockIdx.x * 32;

  // stage 32x256 input rows as bf16 into LDS
  for (int i = tid; i < 32 * FF; i += 256)
    a_lds[i] = f2bf_bits(x_in[row0 * FF + i]);
  __syncthreads();

  // GEMM1: xs = relu(A @ W1 + b1)   (K=256, N=512)
  for (int j = 0; j < 8; ++j) {
    const int n0 = (wn + 4 * j) * 16;
    v8f acc = {};
    for (int k0 = 0; k0 < FF; k0 += 32) {
      v16bf a = load_a_lds(a_lds + wm * 16 * FF, FF, k0, lane);
      v16bf b = load_b_global(w1t, FF, n0, k0, lane);
      acc = wmma_bf16(a, b, acc);
    }
    const int n = n0 + (lane & 15);
    const float bias = b1[n];
    for (int r = 0; r < 8; ++r) {
      const int m = wm * 16 + (lane >> 4) * 8 + r;
      float v = acc[r] + bias;
      v = v > 0.f ? v : 0.f;
      xs_lds[m * HH + n] = f2bf_bits(v);
    }
  }
  __syncthreads();

  // GEMM2: mx = xs @ Wx + bx   (K=512, N=1536)
  for (int j = 0; j < 24; ++j) {
    const int n0 = (wn + 4 * j) * 16;
    v8f acc = {};
    for (int k0 = 0; k0 < HH; k0 += 32) {
      v16bf a = load_a_lds(xs_lds + wm * 16 * HH, HH, k0, lane);
      v16bf b = load_b_global(wxt, HH, n0, k0, lane);
      acc = wmma_bf16(a, b, acc);
    }
    const int n = n0 + (lane & 15);
    const float bias = bx[n];
    for (int r = 0; r < 8; ++r) {
      const int m = wm * 16 + (lane >> 4) * 8 + r;
      mx[(row0 + m) * GG + n] = f2bf_bits(acc[r] + bias);
    }
  }
}

// ---------------------------------------------------------------------------
// Kernel 2: GRU scan over T=512. Batch partitioned: 16 rows per workgroup,
// 16 workgroups, h state lives in LDS (bf16 copy for WMMA-A + f32 copy).
// 32 waves; wave w owns H-columns [16w,16w+16) and the 3 matching gate tiles.
// Wh^T (1.5 MB bf16) streams from L2 every step.
// ---------------------------------------------------------------------------
__global__ __launch_bounds__(1024) void k_gru_scan(
    const unsigned short* __restrict__ wht,  // [1536][512] bf16
    const float* __restrict__ bh,            // [1536]
    const unsigned short* __restrict__ mx,   // [BT][1536] bf16
    float* __restrict__ hT)                  // [256][512] f32
{
  __shared__ __align__(32) unsigned short h_bf[16 * HH];  // 16 KB
  __shared__ __align__(32) float          h_f[16 * HH];   // 32 KB

  const int tid  = threadIdx.x;
  const int lane = tid & 31;
  const int wave = tid >> 5;
  const int b0   = blockIdx.x * 16;

  for (int i = tid; i < 16 * HH; i += 1024) { h_bf[i] = 0; h_f[i] = 0.f; }
  __syncthreads();

  const int n0 = wave * 16;
  const int n  = n0 + (lane & 15);
  const float bhz = bh[n];
  const float bhr = bh[HH + n];
  const float bhh = bh[2 * HH + n];

  for (int t = 0; t < TT; ++t) {
    // prefetch next step's mx rows while we do this step's GEMM
    if (t + 1 < TT) {
      const size_t pr = ((size_t)(b0 + (lane >> 4) * 8) * TT + (t + 1)) * GG;
      __builtin_prefetch(mx + pr + n, 0, 0);
      __builtin_prefetch(mx + pr + HH + n, 0, 0);
      __builtin_prefetch(mx + pr + 2 * HH + n, 0, 0);
    }

    // mh tiles for gates z, r, h: [16 x 16] each, K = 512, shared A-fragment
    v8f az = {}, ar = {}, ah = {};
    for (int k0 = 0; k0 < HH; k0 += 32) {
      v16bf a  = load_a_lds(h_bf, HH, k0, lane);
      v16bf bz = load_b_global(wht, HH, n0, k0, lane);
      v16bf br = load_b_global(wht, HH, HH + n0, k0, lane);
      v16bf bc = load_b_global(wht, HH, 2 * HH + n0, k0, lane);
      az = wmma_bf16(a, bz, az);
      ar = wmma_bf16(a, br, ar);
      ah = wmma_bf16(a, bc, ah);
    }

    float hnew[8];
    for (int r = 0; r < 8; ++r) {
      const int m = (lane >> 4) * 8 + r;
      const size_t mrow = ((size_t)(b0 + m) * TT + t) * GG;
      const float xz = bf2f(mx[mrow + n]);
      const float xr = bf2f(mx[mrow + HH + n]);
      const float xh = bf2f(mx[mrow + 2 * HH + n]);
      const float hz = az[r] + bhz;
      const float hr = ar[r] + bhr;
      const float hh = ah[r] + bhh;
      const float z  = 1.f / (1.f + __expf(-(xz + hz)));
      const float rg = 1.f / (1.f + __expf(-(xr + hr)));
      float cand = xh + rg * hh;
      cand = cand > 0.f ? cand : 0.f;
      const float hold = h_f[m * HH + n];
      hnew[r] = z * hold + (1.f - z) * cand;
    }

    __syncthreads();   // all waves done reading h for this step
    for (int r = 0; r < 8; ++r) {
      const int m = (lane >> 4) * 8 + r;
      h_f[m * HH + n]  = hnew[r];
      h_bf[m * HH + n] = f2bf_bits(hnew[r]);
    }
    __syncthreads();   // h fully updated before next step's GEMM
  }

  for (int i = tid; i < 16 * HH; i += 1024)
    hT[(size_t)b0 * HH + i] = h_f[i];
}

// ---------------------------------------------------------------------------
// Kernel 3: out = [hT, hT] @ W2 + b2  ==  hT @ (W2[:512] + W2[512:]) + b2
// 67 MFLOP -> plain f32 VALU is fine.
// ---------------------------------------------------------------------------
__global__ void k_out_head(const float* __restrict__ hT,
                           const float* __restrict__ w2,
                           const float* __restrict__ b2,
                           float* __restrict__ out) {
  int idx = blockIdx.x * 256 + threadIdx.x;   // 256*128
  if (idx >= BB * 128) return;
  const int b = idx >> 7, j = idx & 127;
  const float* h = hT + (size_t)b * HH;
  float acc = b2[j];
  for (int k = 0; k < HH; ++k)
    acc += h[k] * (w2[k * 128 + j] + w2[(HH + k) * 128 + j]);
  out[idx] = acc;
}

// ---------------------------------------------------------------------------
// Launch
// ---------------------------------------------------------------------------
extern "C" void kernel_launch(void* const* d_in, const int* in_sizes, int n_in,
                              void* d_out, int out_size, void* d_ws, size_t ws_size,
                              hipStream_t stream) {
  (void)in_sizes; (void)n_in; (void)out_size; (void)ws_size;
  const float* inputs = (const float*)d_in[0];
  const float* W1 = (const float*)d_in[1];
  const float* b1 = (const float*)d_in[2];
  const float* Wx = (const float*)d_in[3];
  const float* Wh = (const float*)d_in[4];
  const float* bx = (const float*)d_in[5];
  const float* bh = (const float*)d_in[6];
  const float* W2 = (const float*)d_in[7];
  const float* b2 = (const float*)d_in[8];
  float* out = (float*)d_out;

  // Workspace layout (~406 MB total)
  unsigned short* w1t = (unsigned short*)d_ws;              // 512*256 bf16
  unsigned short* wxt = w1t + (size_t)HH * FF;              // 1536*512 bf16
  unsigned short* wht = wxt + (size_t)GG * HH;              // 1536*512 bf16
  unsigned short* mxs = wht + (size_t)GG * HH;              // BT*1536 bf16
  float* hT = (float*)(mxs + (size_t)BT * GG);              // 256*512 f32

  k_transpose_bf16<<<(HH * FF + 255) / 256, 256, 0, stream>>>(W1, w1t, FF, HH);
  k_transpose_bf16<<<(GG * HH + 255) / 256, 256, 0, stream>>>(Wx, wxt, HH, GG);
  k_transpose_bf16<<<(GG * HH + 255) / 256, 256, 0, stream>>>(Wh, wht, HH, GG);
  k_fused_proj<<<BT / 32, 256, 0, stream>>>(inputs, w1t, b1, wxt, bx, mxs);
  k_gru_scan<<<BB / 16, 1024, 0, stream>>>(wht, bh, mxs, hT);
  k_out_head<<<(BB * 128 + 255) / 256, 256, 0, stream>>>(hT, W2, b2, out);
}
